// MwAN_4011499454717
// MI455X (gfx1250) — compile-verified, matrix operand order checked by reading, not run on hardware
//
#include <hip/hip_runtime.h>

typedef __attribute__((ext_vector_type(16))) _Float16 v16h;
typedef __attribute__((ext_vector_type(8)))  float    v8f;

// Problem dims
#define V_EMB 256
#define E_DIM 128
#define B_SZ  8
#define QL_SZ 128
#define PL_SZ 256
#define AL_SZ 10
#define G3H   384   // 3*H, H=128 for every GRU here

// ---------------------------------------------------------------------------
// CDNA5 async copy to LDS (CDNA5 ISA): per-lane 16B transfer,
// VDST VGPR holds LDS byte offset, VADDR pair holds 64-bit global address.
// Tracked by ASYNCcnt; fence with s_wait_asynccnt then workgroup barrier.
// ---------------------------------------------------------------------------
__device__ __forceinline__ void async_copy_b128(int lds_byte_off, const void* gaddr) {
  asm volatile("global_load_async_to_lds_b128 %0, %1, off"
               :: "v"(lds_byte_off), "v"(gaddr) : "memory");
}
__device__ __forceinline__ void wait_async_cnt() {
  asm volatile("s_wait_asynccnt 0x0" ::: "memory");
}

// ---------------------------------------------------------------------------
// WMMA helpers (gfx1250, wave32). 16-bit A-matrix 16x32 layout per CDNA5 ISA:
// lane holds row M=(lane&15); half h maps to K = (h>>3)*16 + (lane>>4)*8 + (h&7).
// With the (lane>>4)*8 folded into the base pointer, a fragment is two
// contiguous 8-float runs at +0 and +16 -> four float4 loads + cvt.
// ---------------------------------------------------------------------------
__device__ __forceinline__ v8f wmma16(v16h a, v16h b, v8f c) {
  return __builtin_amdgcn_wmma_f32_16x16x32_f16(false, a, false, b, (short)0, c,
                                                false, false);
}

__device__ __forceinline__ v16h cvt_frag(const float* p) {
  float t[16];
  ((float4*)t)[0] = *(const float4*)(p);
  ((float4*)t)[1] = *(const float4*)(p + 4);
  ((float4*)t)[2] = *(const float4*)(p + 16);
  ((float4*)t)[3] = *(const float4*)(p + 20);
  v16h a;
#pragma unroll
  for (int h = 0; h < 16; ++h) a[h] = (_Float16)t[h];
  return a;
}

template <int OP>  // 0: mul, 1: sub
__device__ __forceinline__ v16h op_frag(const float* xp, const float* yp) {
  float x[16], y[16];
  ((float4*)x)[0] = *(const float4*)(xp);
  ((float4*)x)[1] = *(const float4*)(xp + 4);
  ((float4*)x)[2] = *(const float4*)(xp + 16);
  ((float4*)x)[3] = *(const float4*)(xp + 20);
  ((float4*)y)[0] = *(const float4*)(yp);
  ((float4*)y)[1] = *(const float4*)(yp + 4);
  ((float4*)y)[2] = *(const float4*)(yp + 16);
  ((float4*)y)[3] = *(const float4*)(yp + 20);
  v16h a;
#pragma unroll
  for (int h = 0; h < 16; ++h)
    a[h] = (_Float16)(OP ? (x[h] - y[h]) : (x[h] * y[h]));
  return a;
}

// act: 0=none, 1=tanh, 2=leaky_relu(0.01)
__device__ __forceinline__ void store_frag_C(float* C, int ldc, int row0, int n0,
                                             int lane, v8f c, const float* bias,
                                             int act, int M, int N) {
  int n = n0 + (lane & 15);
  if (n >= N) return;
  float bv = bias ? bias[n] : 0.f;
#pragma unroll
  for (int v = 0; v < 8; ++v) {
    int m = row0 + v + ((lane >> 4) << 3);
    if (m < M) {
      float y = c[v] + bv;
      if (act == 1) y = tanhf(y);
      else if (act == 2) y = (y > 0.f) ? y : 0.01f * y;
      C[(long)m * ldc + n] = y;
    }
  }
}

// ---------------------------------------------------------------------------
// Batched GEMM: C[b] = act(A[b]*B[b] + bias). Block = 32x64 macro-tile
// (8 waves, 2x4 of 16x16 WMMA tiles). A and (for TRB) B staged to LDS with
// async-to-LDS b128; fragments built from LDS float4 reads. K % 64 == 0.
// ---------------------------------------------------------------------------
template <bool TRB>
__global__ void gemm_wmma(const float* __restrict__ A, const float* __restrict__ Bm,
                          const float* __restrict__ bias, float* __restrict__ C,
                          int M, int N, int K, int lda, int ldb, int ldc,
                          long sA, long sB, long sC, int act) {
  int batch = blockIdx.y;
  int nMacN = (N + 63) >> 6;
  int bm = blockIdx.x / nMacN, bn = blockIdx.x % nMacN;
  int tid = threadIdx.x;
  int wave = tid >> 5, lane = tid & 31;
  int tmw = wave >> 2, tnw = wave & 3;
  const float* Ab = A + sA * batch;
  const float* Bb = Bm + sB * batch;
  float* Cb = C + sC * batch;

  __shared__ float smem[32 * 64 + 64 * 64];   // As[32][64] | Bs[64][64] (24 KB)
  float* As = smem;
  float* Bs = smem + 32 * 64;
  const int BS_B0 = 32 * 64 * 4;              // byte offset of Bs in LDS

  int ar = tid >> 3, ac = (tid & 7) * 8;      // A staging: 2 x b128 / thread
  int bnr = tid >> 2, bkg = (tid & 3) * 16;   // B staging: 4 x b128 / thread

  int r = lane & 15;
  int hi = (lane >> 4) << 3;
  const float* arow = As + (tmw * 16 + r) * 64;
  const float* brow = Bs + (tnw * 16 + r) * 64;

  v8f acc = {};
  for (int k0 = 0; k0 < K; k0 += 64) {
    __syncthreads();  // previous-iter LDS readers done before restage
    {
      const float* ga = Ab + (long)(bm * 32 + ar) * lda + k0 + ac;
      async_copy_b128((ar * 64 + ac) * 4, ga);
      async_copy_b128((ar * 64 + ac + 4) * 4, ga + 4);
      if (k0 + 64 < K) __builtin_prefetch(ga + 64, 0, 0);
    }
    if (TRB) {
      const float* gb = Bb + (long)(bn * 64 + bnr) * ldb + k0 + bkg;
#pragma unroll
      for (int i = 0; i < 4; ++i)
        async_copy_b128(BS_B0 + (bnr * 64 + bkg + 4 * i) * 4, gb + 4 * i);
    } else {
#pragma unroll
      for (int i = 0; i < 16; ++i)
        Bs[bnr * 64 + bkg + i] = Bb[(long)(k0 + bkg + i) * ldb + (bn * 64 + bnr)];
    }
    wait_async_cnt();
    __syncthreads();
#pragma unroll
    for (int c = 0; c < 2; ++c) {
      int kb = c * 32 + hi;
      v16h af = cvt_frag(arow + kb);
      v16h bf = cvt_frag(brow + kb);
      acc = wmma16(af, bf, acc);
    }
  }
  store_frag_C(Cb, ldc, bm * 32 + tmw * 16, bn * 64 + tnw * 16, lane, acc, bias, act, M, N);
}

// ---------------------------------------------------------------------------
// Fused interaction-attention scores (qtd/qtm/qts):
//   score[b,q,p] = sum_e v[e]*tanh( sum_f W[e,f]*op(X[b,p,f], Y[b,q,f]) )
// Block per (b,q,p-tile16); 8 waves cover e=0..127. Y row and 16x256 X tile
// async-staged to LDS (shared by all waves, full K); K=256 fully unrolled ->
// 8 WMMAs per wave. No [B,Q,P,F] intermediate ever materialized.
// ---------------------------------------------------------------------------
template <int OP>
__global__ void prod_score(const float* __restrict__ X, const float* __restrict__ Y,
                           const float* __restrict__ W, const float* __restrict__ v,
                           float* __restrict__ score, int P, int Q) {
  int ptiles = P >> 4;
  int bidx = blockIdx.x;
  int pt = bidx % ptiles;
  int q = (bidx / ptiles) % Q;
  int b = bidx / (ptiles * Q);
  int tid = threadIdx.x;
  int wave = tid >> 5, lane = tid & 31;

  __shared__ float smem[256 + 16 * 256 + 16];  // Ys | Xs[16][256] | partial
  float* Ys = smem;
  float* Xs = smem + 256;
  float* ssc = smem + 256 + 16 * 256;

  if (tid < 64)
    async_copy_b128(tid * 16, Y + ((long)(b * Q + q)) * 256 + tid * 4);
  {
    int row = tid >> 4, kg = (tid & 15) * 16;
    const float* gx = X + ((long)(b * P + pt * 16 + row)) * 256 + kg;
#pragma unroll
    for (int i = 0; i < 4; ++i)
      async_copy_b128((256 + row * 256 + kg + 4 * i) * 4, gx + 4 * i);
  }
  if (tid < 16) ssc[tid] = 0.f;
  wait_async_cnt();
  __syncthreads();

  int r = lane & 15;
  int hi = (lane >> 4) << 3;
  int n = wave * 16 + (lane & 15);
  const float* xrow = Xs + r * 256;
  const float* Wr = W + (long)n * 256;
  v8f acc = {};
#pragma unroll
  for (int c = 0; c < 8; ++c) {
    int kb = c * 32 + hi;
    v16h af = op_frag<OP>(xrow + kb, Ys + kb);
    v16h bf = cvt_frag(Wr + kb);
    acc = wmma16(af, bf, acc);
  }
  float vn = v[n];
#pragma unroll
  for (int vv = 0; vv < 8; ++vv) {
    int m = vv + ((lane >> 4) << 3);
    atomicAdd(&ssc[m], vn * tanhf(acc[vv]));
  }
  __syncthreads();
  if (tid < 16)
    score[((long)(b * Q + q)) * P + pt * 16 + tid] = ssc[tid];
}

// ---------------------------------------------------------------------------
// Elementwise / small kernels
// ---------------------------------------------------------------------------
__global__ void gather_emb(const float* __restrict__ emb, const int* __restrict__ ids,
                           float* __restrict__ out, int n_tok) {
  int t = blockIdx.x * blockDim.x + threadIdx.x;
  if (t >= n_tok * V_EMB) return;
  int tok = t >> 8, e = t & 255;
  out[t] = emb[(long)ids[tok] * V_EMB + e];
}

// Bidirectional-GRU scan, one direction. pre[b*T+t][384] already = x@Wih^T+bih.
// B=8, H=128 fixed -> 1024 threads; hidden state lives in LDS.
__global__ void gru_scan(const float* __restrict__ pre, const float* __restrict__ Whh,
                         const float* __restrict__ bhh, float* __restrict__ out,
                         int T, int col0, int reverse) {
  int b = threadIdx.x >> 7;
  int j = threadIdx.x & 127;
  __shared__ float h[B_SZ * 128];
  h[threadIdx.x] = 0.f;
  __syncthreads();
  for (int s = 0; s < T; ++s) {
    int tt = reverse ? (T - 1 - s) : s;
    const float* hb = &h[b * 128];
    float gr = 0.f, gz = 0.f, gn = 0.f;
#pragma unroll 4
    for (int i = 0; i < 128; ++i) {
      float hv = hb[i];
      gr += Whh[(long)j * 128 + i] * hv;
      gz += Whh[(long)(128 + j) * 128 + i] * hv;
      gn += Whh[(long)(256 + j) * 128 + i] * hv;
    }
    const float* p = pre + ((long)(b * T + tt)) * G3H;
    float r = 1.f / (1.f + expf(-(p[j] + gr + bhh[j])));
    float z = 1.f / (1.f + expf(-(p[128 + j] + gz + bhh[128 + j])));
    float nn = tanhf(p[256 + j] + r * (gn + bhh[256 + j]));
    float hn = (1.f - z) * nn + z * h[threadIdx.x];
    __syncthreads();
    h[threadIdx.x] = hn;
    __syncthreads();
    out[((long)b * T + tt) * 256 + col0 + j] = hn;
  }
}

// Row softmax, length L<=256, one block per row.
__global__ void softmax_rows(float* __restrict__ buf, int L) {
  int row = blockIdx.x, tid = threadIdx.x;
  float* x = buf + (long)row * L;
  __shared__ float red[256];
  float v = (tid < L) ? x[tid] : -1e30f;
  red[tid] = v;
  __syncthreads();
  for (int s = 128; s > 0; s >>= 1) {
    if (tid < s) red[tid] = fmaxf(red[tid], red[tid + s]);
    __syncthreads();
  }
  float m = red[0];
  __syncthreads();
  float e = (tid < L) ? expf(v - m) : 0.f;
  red[tid] = e;
  __syncthreads();
  for (int s = 128; s > 0; s >>= 1) {
    if (tid < s) red[tid] += red[tid + s];
    __syncthreads();
  }
  if (tid < L) x[tid] = e / red[0];
}

// score[b,q,p] = sum_e vc[e]*tanh(hqc[b,p,e] + hpc[b,q,e])
__global__ void sc_score(const float* __restrict__ hqc, const float* __restrict__ hpc,
                         const float* __restrict__ vc, float* __restrict__ score) {
  int t = blockIdx.x * blockDim.x + threadIdx.x;
  if (t >= B_SZ * QL_SZ * PL_SZ) return;
  int p = t % PL_SZ, q = (t / PL_SZ) % QL_SZ, b = t / (PL_SZ * QL_SZ);
  const float* A = hqc + ((long)(b * PL_SZ + p)) * E_DIM;
  const float* Bv = hpc + ((long)(b * QL_SZ + q)) * E_DIM;
  float s = 0.f;
#pragma unroll 4
  for (int e = 0; e < E_DIM; ++e) s += vc[e] * tanhf(A[e] + Bv[e]);
  score[t] = s;
}

// out[row] = sum_e v[e]*(X[row][e] + extra[row/rows_per_b][e])
__global__ void dot_v(const float* __restrict__ X, const float* __restrict__ v,
                      const float* __restrict__ extra, int rows_per_b,
                      float* __restrict__ out, int E, int rows) {
  int row = blockIdx.x * blockDim.x + threadIdx.x;
  if (row >= rows) return;
  const float* x = X + (long)row * E;
  const float* ex = extra ? extra + (long)(row / rows_per_b) * E : nullptr;
  float s = 0.f;
  for (int e = 0; e < E; ++e) s += v[e] * (x[e] + (ex ? ex[e] : 0.f));
  out[row] = s;
}

// out[b][e] = sum_l w[b][l] * X[b][l][e]
__global__ void wsum(const float* __restrict__ w, const float* __restrict__ X,
                     float* __restrict__ out, int L, int E) {
  int t = blockIdx.x * blockDim.x + threadIdx.x;
  if (t >= B_SZ * E) return;
  int b = t / E, e = t % E;
  float s = 0.f;
  for (int l = 0; l < L; ++l) s += w[b * L + l] * X[((long)b * L + l) * E + e];
  out[t] = s;
}

__global__ void copy_to_agg(const float* __restrict__ hp, float* __restrict__ agg) {
  int t = blockIdx.x * blockDim.x + threadIdx.x;
  if (t >= B_SZ * QL_SZ * 256) return;
  int row = t >> 8, e = t & 255;
  agg[(long)row * 1536 + e] = hp[t];
}

// Candidate pooling: a_emb[b][c][:] = sum_t softmax_t(h.a_att)[t] * h[t][:]
__global__ void cand_att(const float* __restrict__ h, const float* __restrict__ a_att,
                         float* __restrict__ a_emb) {
  int idx = blockIdx.x;        // c*8 + b
  int c = idx >> 3, b = idx & 7;
  int tid = threadIdx.x;
  __shared__ float red[256];
  __shared__ float s[AL_SZ];
  const float* hb = h + ((long)(c * B_SZ + b) * AL_SZ) * 256;
  for (int t = 0; t < AL_SZ; ++t) {
    red[tid] = hb[t * 256 + tid] * a_att[tid];
    __syncthreads();
    for (int st = 128; st > 0; st >>= 1) {
      if (tid < st) red[tid] += red[tid + st];
      __syncthreads();
    }
    if (tid == 0) s[t] = red[0];
    __syncthreads();
  }
  if (tid == 0) {
    float m = -1e30f, sum = 0.f;
    for (int t = 0; t < AL_SZ; ++t) m = fmaxf(m, s[t]);
    for (int t = 0; t < AL_SZ; ++t) { s[t] = expf(s[t] - m); sum += s[t]; }
    for (int t = 0; t < AL_SZ; ++t) s[t] /= sum;
  }
  __syncthreads();
  float acc = 0.f;
  for (int t = 0; t < AL_SZ; ++t) acc += s[t] * hb[t * 256 + tid];
  a_emb[((long)b * 3 + c) * 256 + tid] = acc;
}

__global__ void logits_k(const float* __restrict__ a_emb, const float* __restrict__ enc,
                         float* __restrict__ out) {
  int t = threadIdx.x;
  if (t >= 24) return;
  int b = t / 3, c = t % 3;
  float s = 0.f;
  for (int e = 0; e < 256; ++e)
    s += a_emb[((long)b * 3 + c) * 256 + e] * enc[b * 256 + e];
  out[t] = s;
}

// ---------------------------------------------------------------------------
// Host-side launch plumbing
// ---------------------------------------------------------------------------
static void gemm(hipStream_t st, bool trb, const float* A, const float* B,
                 const float* bias, float* C, int M, int N, int K, int lda,
                 int ldb, int ldc, long sA, long sB, long sC, int batch, int act) {
  int gm = (M + 31) / 32, gn = (N + 63) / 64;
  dim3 grid(gm * gn, batch);
  if (trb)
    gemm_wmma<true><<<grid, 256, 0, st>>>(A, B, bias, C, M, N, K, lda, ldb, ldc, sA, sB, sC, act);
  else
    gemm_wmma<false><<<grid, 256, 0, st>>>(A, B, bias, C, M, N, K, lda, ldb, ldc, sA, sB, sC, act);
}

extern "C" void kernel_launch(void* const* d_in, const int* in_sizes, int n_in,
                              void* d_out, int out_size, void* d_ws, size_t ws_size,
                              hipStream_t stream) {
  (void)in_sizes; (void)n_in; (void)out_size; (void)ws_size;
  const float* emb   = (const float*)d_in[0];
  const float* a_att = (const float*)d_in[1];
  const float* Wc1 = (const float*)d_in[2];
  const float* Wc2 = (const float*)d_in[3];
  const float* vc  = (const float*)d_in[4];
  const float* Wb  = (const float*)d_in[5];
  const float* Wd  = (const float*)d_in[6];
  const float* vd  = (const float*)d_in[7];
  const float* Wm  = (const float*)d_in[8];
  const float* vm  = (const float*)d_in[9];
  const float* Ws  = (const float*)d_in[10];
  const float* vs  = (const float*)d_in[11];
  const float* Wq  = (const float*)d_in[12];
  const float* vq  = (const float*)d_in[13];
  const float* Wp1 = (const float*)d_in[14];
  const float* Wp2 = (const float*)d_in[15];
  const float* vp  = (const float*)d_in[16];
  const float* Wpred = (const float*)d_in[17];
  const float* q_Wih = (const float*)d_in[18];
  const float* q_Whh = (const float*)d_in[19];
  const float* q_bih = (const float*)d_in[20];
  const float* q_bhh = (const float*)d_in[21];
  const float* p_Wih = (const float*)d_in[22];
  const float* p_Whh = (const float*)d_in[23];
  const float* p_bih = (const float*)d_in[24];
  const float* p_bhh = (const float*)d_in[25];
  const float* a_Wih = (const float*)d_in[26];
  const float* a_Whh = (const float*)d_in[27];
  const float* a_bih = (const float*)d_in[28];
  const float* a_bhh = (const float*)d_in[29];
  const float* g_Wih = (const float*)d_in[30];
  const float* g_Whh = (const float*)d_in[31];
  const float* g_bih = (const float*)d_in[32];
  const float* g_bhh = (const float*)d_in[33];
  const int* query   = (const int*)d_in[34];
  const int* passage = (const int*)d_in[35];
  const int* cand[3] = {(const int*)d_in[36], (const int*)d_in[37], (const int*)d_in[38]};
  float* out = (float*)d_out;

  // workspace layout (floats)
  float* Wsp = (float*)d_ws;
  size_t o = 0;
  auto alloc = [&](size_t n) { float* p = Wsp + o; o += n; return p; };
  float* X_P   = alloc(8L * 256 * 256);   // emb[passage]
  float* X_Q   = alloc(8L * 128 * 256);   // emb[query]
  float* X_C   = alloc(3L * 8 * 10 * 256);
  float* HQ    = alloc(8L * 256 * 256);
  float* HP    = alloc(8L * 128 * 256);
  float* HC    = alloc(3L * 8 * 10 * 256);
  float* PRE_F = alloc(8L * 256 * 384);
  float* PRE_B = alloc(8L * 256 * 384);
  float* SCORE = alloc(8L * 128 * 256);
  float* HQ_C1 = alloc(8L * 256 * 128);
  float* HP_C2 = alloc(8L * 128 * 128);
  float* HQ_B  = alloc(8L * 256 * 256);
  float* HQ_Q  = alloc(8L * 256 * 128);
  float* AGG   = alloc(8L * 128 * 1536);
  float* AGR   = alloc(8L * 128 * 256);
  float* AGP1  = alloc(8L * 128 * 128);
  float* RQ    = alloc(8L * 256);
  float* RQW   = alloc(8L * 128);
  float* RP    = alloc(8L * 256);
  float* ENC   = alloc(8L * 256);
  float* AEMB  = alloc(8L * 3 * 256);
  float* SC2   = alloc(4096);
  float* SCQ = SC2;          // [8][256]
  float* SCP = SC2 + 2048;   // [8][128]

  // ---- embedding gathers ----
  gather_emb<<<(8 * 256 * 256 + 255) / 256, 256, 0, stream>>>(emb, passage, X_P, 8 * 256);
  gather_emb<<<(8 * 128 * 256 + 255) / 256, 256, 0, stream>>>(emb, query, X_Q, 8 * 128);
  for (int c = 0; c < 3; ++c)
    gather_emb<<<(8 * 10 * 256 + 255) / 256, 256, 0, stream>>>(emb, cand[c], X_C + c * 20480L, 80);

  // ---- hq = bigru(emb[passage], q_*) : T=256 ----
  gemm(stream, true, X_P, q_Wih, q_bih, PRE_F, 2048, 384, 256, 256, 256, 384, 0, 0, 0, 1, 0);
  gemm(stream, true, X_P, q_Wih + 384L * 256, q_bih + 384, PRE_B, 2048, 384, 256, 256, 256, 384, 0, 0, 0, 1, 0);
  gru_scan<<<1, 1024, 0, stream>>>(PRE_F, q_Whh, q_bhh, HQ, 256, 0, 0);
  gru_scan<<<1, 1024, 0, stream>>>(PRE_B, q_Whh + 384L * 128, q_bhh + 384, HQ, 256, 128, 1);

  // ---- hp = bigru(emb[query], p_*) : T=128 ----
  gemm(stream, true, X_Q, p_Wih, p_bih, PRE_F, 1024, 384, 256, 256, 256, 384, 0, 0, 0, 1, 0);
  gemm(stream, true, X_Q, p_Wih + 384L * 256, p_bih + 384, PRE_B, 1024, 384, 256, 256, 256, 384, 0, 0, 0, 1, 0);
  gru_scan<<<1, 1024, 0, stream>>>(PRE_F, p_Whh, p_bhh, HP, 128, 0, 0);
  gru_scan<<<1, 1024, 0, stream>>>(PRE_B, p_Whh + 384L * 128, p_bhh + 384, HP, 128, 128, 1);

  // ---- candidate encoders : T=10 ----
  for (int c = 0; c < 3; ++c) {
    const float* xc = X_C + c * 20480L;
    float* hc = HC + c * 20480L;
    gemm(stream, true, xc, a_Wih, a_bih, PRE_F, 80, 384, 256, 256, 256, 384, 0, 0, 0, 1, 0);
    gemm(stream, true, xc, a_Wih + 384L * 256, a_bih + 384, PRE_B, 80, 384, 256, 256, 256, 384, 0, 0, 0, 1, 0);
    gru_scan<<<1, 1024, 0, stream>>>(PRE_F, a_Whh, a_bhh, hc, 10, 0, 0);
    gru_scan<<<1, 1024, 0, stream>>>(PRE_B, a_Whh + 384L * 128, a_bhh + 384, hc, 10, 128, 1);
  }
  cand_att<<<24, 256, 0, stream>>>(HC, a_att, AEMB);

  // ---- shared projections ----
  gemm(stream, true, HQ, Wc1, nullptr, HQ_C1, 2048, 128, 256, 256, 256, 128, 0, 0, 0, 1, 0);
  gemm(stream, true, HP, Wc2, nullptr, HP_C2, 1024, 128, 256, 256, 256, 128, 0, 0, 0, 1, 0);
  gemm(stream, true, HQ, Wb, nullptr, HQ_B, 2048, 256, 256, 256, 256, 256, 0, 0, 0, 1, 0);
  gemm(stream, true, HQ, Wq, nullptr, HQ_Q, 2048, 128, 256, 256, 256, 128, 0, 0, 0, 1, 1); // tanh

  const long sQ = 128L * 256, sP = 256L * 256, sAg = 128L * 1536, sS128 = 128L * 128;

  // ---- concat attention (qtc -> agg[512:768]) ----
  sc_score<<<(8 * 128 * 256 + 255) / 256, 256, 0, stream>>>(HQ_C1, HP_C2, vc, SCORE);
  softmax_rows<<<1024, 256, 0, stream>>>(SCORE, 256);
  gemm(stream, false, SCORE, HQ, nullptr, AGG + 512, 128, 256, 256, 256, 256, 1536, sQ, sP, sAg, 8, 0);

  // ---- bilinear attention (qtb -> agg[1024:1280]) ----
  gemm(stream, true, HP, HQ_B, nullptr, SCORE, 128, 256, 256, 256, 256, 256, sQ, sP, sQ, 8, 0);
  softmax_rows<<<1024, 256, 0, stream>>>(SCORE, 256);
  gemm(stream, false, SCORE, HQ, nullptr, AGG + 1024, 128, 256, 256, 256, 256, 1536, sQ, sP, sAg, 8, 0);

  // ---- dot attention (qtd -> agg[768:1024]) ----
  prod_score<0><<<8 * 128 * 16, 256, 0, stream>>>(HQ, HP, Wd, vd, SCORE, 256, 128);
  softmax_rows<<<1024, 256, 0, stream>>>(SCORE, 256);
  gemm(stream, false, SCORE, HQ, nullptr, AGG + 768, 128, 256, 256, 256, 256, 1536, sQ, sP, sAg, 8, 0);

  // ---- minus attention (qtm -> agg[1280:1536]) ----
  prod_score<1><<<8 * 128 * 16, 256, 0, stream>>>(HQ, HP, Wm, vm, SCORE, 256, 128);
  softmax_rows<<<1024, 256, 0, stream>>>(SCORE, 256);
  gemm(stream, false, SCORE, HQ, nullptr, AGG + 1280, 128, 256, 256, 256, 256, 1536, sQ, sP, sAg, 8, 0);

  // ---- self attention on hp (qts -> agg[256:512]) ----
  prod_score<0><<<8 * 128 * 8, 256, 0, stream>>>(HP, HP, Ws, vs, SCORE, 128, 128);
  softmax_rows<<<1024, 256, 0, stream>>>(SCORE, 128);
  gemm(stream, false, SCORE, HP, nullptr, AGG + 256, 128, 256, 128, 128, 256, 1536, sS128, sQ, sAg, 8, 0);

  // ---- agg[0:256] = hp ----
  copy_to_agg<<<(8 * 128 * 256 + 255) / 256, 256, 0, stream>>>(HP, AGG);

  // ---- agg_rep = bigru(agg, g_*) : T=128, input 1536 ----
  gemm(stream, true, AGG, g_Wih, g_bih, PRE_F, 1024, 384, 1536, 1536, 1536, 384, 0, 0, 0, 1, 0);
  gemm(stream, true, AGG, g_Wih + 384L * 1536, g_bih + 384, PRE_B, 1024, 384, 1536, 1536, 1536, 384, 0, 0, 0, 1, 0);
  gru_scan<<<1, 1024, 0, stream>>>(PRE_F, g_Whh, g_bhh, AGR, 128, 0, 0);
  gru_scan<<<1, 1024, 0, stream>>>(PRE_B, g_Whh + 384L * 128, g_bhh + 384, AGR, 128, 128, 1);

  // ---- rq pooling over hq ----
  dot_v<<<(2048 + 255) / 256, 256, 0, stream>>>(HQ_Q, vq, nullptr, 1, SCQ, 128, 2048);
  softmax_rows<<<8, 256, 0, stream>>>(SCQ, 256);
  wsum<<<(8 * 256 + 255) / 256, 256, 0, stream>>>(SCQ, HQ, RQ, 256, 256);

  // ---- rp pooling over agg_rep ----
  gemm(stream, true, AGR, Wp1, nullptr, AGP1, 1024, 128, 256, 256, 256, 128, 0, 0, 0, 1, 0);
  gemm(stream, true, RQ, Wp2, nullptr, RQW, 8, 128, 256, 256, 256, 128, 0, 0, 0, 1, 0);
  dot_v<<<(1024 + 255) / 256, 256, 0, stream>>>(AGP1, vp, RQW, 128, SCP, 128, 1024);
  softmax_rows<<<8, 256, 0, stream>>>(SCP, 128);
  wsum<<<(8 * 256 + 255) / 256, 256, 0, stream>>>(SCP, AGR, RP, 128, 256);

  // ---- classifier head ----
  gemm(stream, true, RP, Wpred, nullptr, ENC, 8, 256, 256, 256, 256, 256, 0, 0, 0, 1, 2);
  logits_k<<<1, 32, 0, stream>>>(AEMB, ENC, out);
}
